// DeepCFRModel_89739046682936
// MI455X (gfx1250) — matrix operand order repaired and checked by reading, not compiled, not run on hardware
//
#include <hip/hip_runtime.h>
#include <hip/hip_bf16.h>
#include <math.h>

typedef __bf16 bf16;
typedef __attribute__((ext_vector_type(16))) __bf16 v16bf;
typedef __attribute__((ext_vector_type(8)))  __bf16 bf16x8;
typedef __attribute__((ext_vector_type(8)))  float  v8f;

static constexpr int NB    = 32768;   // batch
static constexpr int KFEAT = 1865;    // 7*256 + 9 + 64
static constexpr int KPADF = 1888;    // padded to mult of 32

__device__ inline float gelu_f(float x) {
  return 0.5f * x * (1.0f + erff(x * 0.7071067811865475f));
}

// ---- WMMA fragment helpers -------------------------------------------------
__device__ inline v16bf make_afrag(const bf16* rowptr) {
  // rowptr = &X[row][kk + 8*(lane>>4)]; two contiguous 8-elem runs (16B each)
  bf16x8 lo = *(const bf16x8*)(rowptr);
  bf16x8 hi = *(const bf16x8*)(rowptr + 16);
  v16bf a;
#pragma unroll
  for (int i = 0; i < 8; i++) { a[i] = lo[i]; a[i + 8] = hi[i]; }
  return a;
}

__device__ inline v16bf load_bfrag(const bf16* __restrict__ WT, int ntGlobal,
                                   int kt, int Ktiles, int lane) {
  return *(const v16bf*)(WT + (((size_t)ntGlobal * Ktiles + kt) << 9) + (lane << 4));
}

__device__ inline v8f wmma_bf16(v16bf a, v16bf b, v8f c) {
  return __builtin_amdgcn_wmma_f32_16x16x32_bf16(false, a, false, b, (short)0, c,
                                                 false, false);
}

// ---- weight prep: fp32 (K x N) row-major -> tiled bf16 B-fragment layout ---
// tile (ntile, kt): 512 bf16 = 32 lanes x 16 elems; elem: N=ntile*16+(lane&15),
// K = kt*32 + 8*(lane>>4) + (e&7) + 16*(e>>3).  mode 1: hW1 (4,256,128) viewed as 256x512.
__global__ __launch_bounds__(256) void prep_w(const float* __restrict__ src,
                                              const float* __restrict__ srcB,
                                              int K, int N, int Ktiles,
                                              bf16* __restrict__ dst, int mode) {
  size_t idx = (size_t)blockIdx.x * 256 + threadIdx.x;
  size_t total = (size_t)Ktiles * (N >> 4) * 512;
  if (idx >= total) return;
  int within = (int)(idx & 511);
  size_t tidx = idx >> 9;
  int lane = within >> 4, e = within & 15;
  int kt = (int)(tidx % Ktiles);
  int ntile = (int)(tidx / Ktiles);
  int n = ntile * 16 + (lane & 15);
  int k = kt * 32 + (lane >> 4) * 8 + (e & 7) + ((e >> 3) << 4);
  float v = 0.0f;
  if (k < K) {
    if (mode == 1) {
      int s = n >> 7, j = n & 127;
      v = src[((size_t)(s * 256 + k)) * 128 + j];
    } else {
      v = src[(size_t)k * N + n];
      if (srcB) v += srcB[(size_t)k * N + n];
    }
  }
  dst[idx] = (bf16)v;
}

// ---- generic bf16 GEMM: out(MxN fp32) = X(M x Kpad bf16) @ Wtiled + bias ---
// 8 waves/block; wave w -> rows blockIdx.x*64+(w&3)*16, cols blockIdx.y*256+(w>>2)*128
// Each wave: 16x128 output tile = 8 f32 accumulators; one A-frag feeds 8 WMMAs.
// epi: 0 none, 1 sigmoid, 2 tanh.  res: optional bf16 residual (ld = ldres).
__global__ __launch_bounds__(256) void gemm_bf16(const bf16* __restrict__ X, int Kpad,
                                                 const bf16* __restrict__ WT,
                                                 const float* __restrict__ bias,
                                                 const bf16* __restrict__ res, int ldres,
                                                 float* __restrict__ out, int N, int epi) {
  int w = threadIdx.x >> 5, lane = threadIdx.x & 31;
  int rowBase = blockIdx.x * 64 + (w & 3) * 16;
  int colBase = blockIdx.y * 256 + (w >> 2) * 128;
  int Ktiles = Kpad >> 5;
  v8f acc[8];
#pragma unroll
  for (int t = 0; t < 8; t++) acc[t] = v8f{0.f, 0.f, 0.f, 0.f, 0.f, 0.f, 0.f, 0.f};

  const bf16* xbase =
      X + (size_t)(rowBase + (lane & 15)) * Kpad + ((lane >> 4) << 3);
  const bf16* wbase =
      WT + (((size_t)(colBase >> 4) * Ktiles) << 9) + (lane << 4);
  for (int kt = 0; kt < Ktiles; kt++) {
    v16bf A = make_afrag(xbase + (kt << 5));
#pragma unroll
    for (int t = 0; t < 8; t++) {
      v16bf Bf = *(const v16bf*)(wbase + (((size_t)t * Ktiles + kt) << 9));
      acc[t] = wmma_bf16(A, Bf, acc[t]);
    }
  }
  int rsub = (lane >> 4) * 8;
#pragma unroll
  for (int t = 0; t < 8; t++) {
    int col = colBase + t * 16 + (lane & 15);
    float bv = bias ? bias[col] : 0.0f;
#pragma unroll
    for (int r = 0; r < 8; r++) {
      size_t row = (size_t)rowBase + rsub + r;
      float v = acc[t][r] + bv;
      if (res) v += (float)res[row * ldres + col];
      if (epi == 1) v = 1.0f / (1.0f + expf(-v));
      else if (epi == 2) v = tanhf(v);
      out[row * N + col] = v;
    }
  }
}

// ---- LayerNorm + GELU: fp32 in (row-major N), bf16 out (ld = ldo) ----------
__global__ __launch_bounds__(256) void ln_gelu(const float* __restrict__ in, int N,
                                               const float* __restrict__ g,
                                               const float* __restrict__ b,
                                               bf16* __restrict__ out, int ldo) {
  int row = blockIdx.x;
  int tid = threadIdx.x, w = tid >> 5, lane = tid & 31;
  const float* x = in + (size_t)row * N;
  float s = 0.f, s2 = 0.f;
  for (int c = tid; c < N; c += 256) { float v = x[c]; s += v; s2 += v * v; }
#pragma unroll
  for (int off = 16; off > 0; off >>= 1) {
    s += __shfl_xor(s, off, 32);
    s2 += __shfl_xor(s2, off, 32);
  }
  __shared__ float red[16];
  if (lane == 0) { red[w] = s; red[8 + w] = s2; }
  __syncthreads();
  float ts = 0.f, ts2 = 0.f;
#pragma unroll
  for (int i = 0; i < 8; i++) { ts += red[i]; ts2 += red[8 + i]; }
  float m = ts / (float)N;
  float var = ts2 / (float)N - m * m;
  float inv = rsqrtf(var + 1e-5f);
  for (int c = tid; c < N; c += 256) {
    float v = (x[c] - m) * inv * g[c] + b[c];
    out[(size_t)row * ldo + c] = (bf16)gelu_f(v);
  }
}

// ---- fused embedding + MHA + LN (4 batch rows / block) ---------------------
// token-major tile: rti = tok*4 + r, 28 real rows padded to 32 (2 M-tiles).
__global__ __launch_bounds__(256) void attn_kernel(
    const int* __restrict__ priv, const int* __restrict__ pub,
    const float* __restrict__ table,
    const bf16* __restrict__ WqT, const bf16* __restrict__ WkT,
    const bf16* __restrict__ WvT, const bf16* __restrict__ WoT,
    const float* __restrict__ bq, const float* __restrict__ bk,
    const float* __restrict__ bv, const float* __restrict__ bo,
    const float* __restrict__ lng, const float* __restrict__ lnb,
    bf16* __restrict__ featbf) {
  __shared__ __align__(16) char smem[82048];
  bf16* Xs = (bf16*)smem;                   // 32x256 bf16 (16KB)
  bf16* Qs = (bf16*)(smem + 16384);
  bf16* Ks = (bf16*)(smem + 32768);
  bf16* Vs = (bf16*)(smem + 49152);
  bf16* As = (bf16*)(smem + 65536);         // attention output (16KB)
  float* Os = (float*)(smem + 16384);       // aliases Qs+Ks after attention (32KB)
  int* pads = (int*)(smem + 81920);         // 32 ints

  int tid = threadIdx.x;
  int b0 = blockIdx.x * 4;

  // stage tokens: embedding + positional encoding -> bf16 LDS
  for (int idx = tid; idx < 28 * 256; idx += 256) {
    int rti = idx >> 8, c = idx & 255;
    int tok = rti >> 2, r = rti & 3;
    int row = b0 + r;
    int card = (tok < 2) ? priv[row * 2 + tok] : pub[row * 5 + tok - 2];
    float v = 0.0f;
    if (card >= 0) v = table[(size_t)card * 256 + c];
    if (tok >= 2) {
      float pos = (float)(tok - 2);
      float dv = expf(-9.210340371976184f * (float)(c & ~1) * (1.0f / 256.0f));
      v += (c & 1) ? cosf(pos * dv) : sinf(pos * dv);
    }
    Xs[rti * 256 + c] = (bf16)v;
    if (c == 0) pads[rti] = (card < 0) ? 1 : 0;
  }
  for (int idx = tid; idx < 4 * 256; idx += 256) Xs[28 * 256 + idx] = (bf16)0.0f;
  __syncthreads();

  int w = tid >> 5, lane = tid & 31;
  int colBase = w * 32;  // wave covers 32 output cols

  // Q/K/V projections (WMMA)
  {
    const bf16* Wm[3] = {WqT, WkT, WvT};
    const float* Bm[3] = {bq, bk, bv};
    bf16* Dm[3] = {Qs, Ks, Vs};
    for (int m = 0; m < 3; m++) {
      for (int mt = 0; mt < 2; mt++) {
        v8f a0 = v8f{0.f,0.f,0.f,0.f,0.f,0.f,0.f,0.f};
        v8f a1 = a0;
        const bf16* abase = Xs + (mt * 16 + (lane & 15)) * 256 + ((lane >> 4) << 3);
        for (int kt = 0; kt < 8; kt++) {
          v16bf A = make_afrag(abase + (kt << 5));
          v16bf B0 = load_bfrag(Wm[m], (colBase >> 4), kt, 8, lane);
          v16bf B1 = load_bfrag(Wm[m], (colBase >> 4) + 1, kt, 8, lane);
          a0 = wmma_bf16(A, B0, a0);
          a1 = wmma_bf16(A, B1, a1);
        }
        int rowb = mt * 16 + (lane >> 4) * 8;
        int c0 = colBase + (lane & 15);
#pragma unroll
        for (int r = 0; r < 8; r++) {
          Dm[m][(rowb + r) * 256 + c0] = (bf16)(a0[r] + Bm[m][c0]);
          Dm[m][(rowb + r) * 256 + c0 + 16] = (bf16)(a1[r] + Bm[m][c0 + 16]);
        }
      }
    }
  }
  __syncthreads();

  // scaled-dot attention per (row, head, query): 7 keys, head_dim 32 (VALU)
  if (tid < 224) {
    int r = tid / 56, rem = tid % 56, h = rem / 7, qi = rem % 7;
    float q[32];
#pragma unroll
    for (int d = 0; d < 32; d++) q[d] = (float)Qs[(qi * 4 + r) * 256 + h * 32 + d];
    float sc[7];
    for (int j = 0; j < 7; j++) {
      if (pads[j * 4 + r]) { sc[j] = -1e9f; continue; }
      float acc = 0.f;
#pragma unroll
      for (int d = 0; d < 32; d++) acc += q[d] * (float)Ks[(j * 4 + r) * 256 + h * 32 + d];
      sc[j] = acc * 0.17677669529663687f;  // 1/sqrt(32)
    }
    float mx = sc[0];
    for (int j = 1; j < 7; j++) mx = fmaxf(mx, sc[j]);
    float ssum = 0.f;
    for (int j = 0; j < 7; j++) { sc[j] = expf(sc[j] - mx); ssum += sc[j]; }
    float invs = 1.0f / ssum;
    float o[32];
#pragma unroll
    for (int d = 0; d < 32; d++) o[d] = 0.f;
    for (int j = 0; j < 7; j++) {
      float a = sc[j] * invs;
#pragma unroll
      for (int d = 0; d < 32; d++) o[d] += a * (float)Vs[(j * 4 + r) * 256 + h * 32 + d];
    }
#pragma unroll
    for (int d = 0; d < 32; d++) As[(qi * 4 + r) * 256 + h * 32 + d] = (bf16)o[d];
  } else {
    int t2 = tid - 224;
    for (int i = t2; i < 4 * 256; i += 32) As[28 * 256 + i] = (bf16)0.0f;
  }
  __syncthreads();

  // output projection + residual -> Os (fp32, aliases Qs/Ks)
  for (int mt = 0; mt < 2; mt++) {
    v8f a0 = v8f{0.f,0.f,0.f,0.f,0.f,0.f,0.f,0.f};
    v8f a1 = a0;
    const bf16* abase = As + (mt * 16 + (lane & 15)) * 256 + ((lane >> 4) << 3);
    for (int kt = 0; kt < 8; kt++) {
      v16bf A = make_afrag(abase + (kt << 5));
      v16bf B0 = load_bfrag(WoT, (colBase >> 4), kt, 8, lane);
      v16bf B1 = load_bfrag(WoT, (colBase >> 4) + 1, kt, 8, lane);
      a0 = wmma_bf16(A, B0, a0);
      a1 = wmma_bf16(A, B1, a1);
    }
    int rowb = mt * 16 + (lane >> 4) * 8;
    int c0 = colBase + (lane & 15);
#pragma unroll
    for (int r = 0; r < 8; r++) {
      int rti = rowb + r;
      Os[rti * 256 + c0] = a0[r] + bo[c0] + (float)Xs[rti * 256 + c0];
      Os[rti * 256 + c0 + 16] = a1[r] + bo[c0 + 16] + (float)Xs[rti * 256 + c0 + 16];
    }
  }
  __syncthreads();

  // per-token LayerNorm -> feat_in cols [tok*256, +255]; 8 threads per token row
  if (tid < 224) {
    int rti = tid >> 3;   // 0..27
    int sub = tid & 7;
    float s = 0.f, s2 = 0.f;
    for (int c = sub; c < 256; c += 8) {
      float v = Os[rti * 256 + c];
      s += v; s2 += v * v;
    }
#pragma unroll
    for (int off = 1; off < 8; off <<= 1) {  // groups of 8 stay inside a wave32
      s += __shfl_xor(s, off, 32);
      s2 += __shfl_xor(s2, off, 32);
    }
    float m = s * (1.0f / 256.0f);
    float var = s2 * (1.0f / 256.0f) - m * m;
    float inv = rsqrtf(var + 1e-5f);
    int tok = rti >> 2, r = rti & 3;
    int row = b0 + r;
    bf16* dst = featbf + (size_t)row * KPADF + tok * 256;
    for (int c = sub * 32; c < sub * 32 + 32; c += 2) {
      float v0 = (Os[rti * 256 + c] - m) * inv * lng[c] + lnb[c];
      float v1 = (Os[rti * 256 + c + 1] - m) * inv * lng[c + 1] + lnb[c + 1];
      bf16 h0 = (bf16)v0, h1 = (bf16)v1;
      unsigned int pk = ((unsigned int)(*(unsigned short*)&h1) << 16) |
                        (unsigned int)(*(unsigned short*)&h0);
      *(unsigned int*)(dst + c) = pk;
    }
  }
}

// ---- scalar feature columns of feat_in (cols 1792..1887, incl zero pad) ----
__global__ __launch_bounds__(256) void smallfeat_kernel(
    const float* __restrict__ stacks, const float* __restrict__ bets,
    const float* __restrict__ act, const int* __restrict__ stage,
    const int* __restrict__ pos, const float* __restrict__ st_emb,
    const float* __restrict__ p_emb, bf16* __restrict__ featbf) {
  int idx = blockIdx.x * 256 + threadIdx.x;
  if (idx >= NB * 96) return;
  int row = idx / 96, c = idx % 96;
  float v = 0.f;
  if (c < 3) v = stacks[row * 3 + c];
  else if (c < 6) v = bets[row * 3 + c - 3];
  else if (c < 9) v = act[row * 3 + c - 6];
  else if (c < 41) v = st_emb[stage[row] * 32 + (c - 9)];
  else if (c < 73) v = p_emb[pos[row] * 32 + (c - 41)];
  featbf[(size_t)row * KPADF + 1792 + c] = (bf16)v;
}

__global__ __launch_bounds__(256) void cvtprev_kernel(const float* __restrict__ prev,
                                                      bf16* __restrict__ comb) {
  int idx = blockIdx.x * 256 + threadIdx.x;
  int row = idx >> 8, c = idx & 255;
  comb[((size_t)row << 9) + 256 + c] = (bf16)prev[idx];
}

__global__ __launch_bounds__(256) void mulrp_kernel(const float* __restrict__ rb,
                                                    const float* __restrict__ prev,
                                                    bf16* __restrict__ comb) {
  int idx = blockIdx.x * 256 + threadIdx.x;
  int row = idx >> 8, c = idx & 255;
  comb[((size_t)row << 9) + 256 + c] = (bf16)(rb[idx] * prev[idx]);
}

__global__ __launch_bounds__(256) void gruout_kernel(const float* __restrict__ zb,
                                                     const float* __restrict__ cb,
                                                     const float* __restrict__ prev,
                                                     float* __restrict__ ctx_out,
                                                     bf16* __restrict__ comb) {
  int idx = blockIdx.x * 256 + threadIdx.x;
  int row = idx >> 8, c = idx & 255;
  float z = zb[idx], cd = cb[idx], p = prev[idx];
  float nc = (1.0f - z) * p + z * cd;
  ctx_out[idx] = nc;
  comb[((size_t)row << 9) + 256 + c] = (bf16)nc;
}

// ---- per-stage heads: gelu(t6[s-slice]+hb1[s]) @ hW2[s] + hb2[s], softmax --
__global__ __launch_bounds__(256) void head_kernel(const float* __restrict__ t6,
                                                   const int* __restrict__ stage,
                                                   const float* __restrict__ hb1,
                                                   const float* __restrict__ hW2,
                                                   const float* __restrict__ hb2,
                                                   const float* __restrict__ amask,
                                                   float* __restrict__ probs) {
  int row = blockIdx.x * 256 + threadIdx.x;
  int s = stage[row];
  const float* trow = t6 + (size_t)row * 512 + s * 128;
  const float* b1 = hb1 + s * 128;
  const float* w2 = hW2 + (size_t)s * 128 * 10;
  float l[10];
#pragma unroll
  for (int a = 0; a < 10; a++) l[a] = hb2[s * 10 + a];
  for (int j = 0; j < 128; j++) {
    float hj = gelu_f(trow[j] + b1[j]);
#pragma unroll
    for (int a = 0; a < 10; a++) l[a] += hj * w2[j * 10 + a];
  }
  float mx = l[0];
#pragma unroll
  for (int a = 1; a < 10; a++) mx = fmaxf(mx, l[a]);
  float e[10], sum = 0.f, msum = 0.f;
#pragma unroll
  for (int a = 0; a < 10; a++) {
    float mk = amask[(size_t)row * 10 + a];
    e[a] = expf(l[a] - mx) * mk;
    sum += e[a];
    msum += mk;
  }
  if (sum < 1e-12f) {
#pragma unroll
    for (int a = 0; a < 10; a++) e[a] = amask[(size_t)row * 10 + a];
    sum = msum;
  }
  float inv = 1.0f / sum;
#pragma unroll
  for (int a = 0; a < 10; a++) probs[(size_t)row * 10 + a] = e[a] * inv;
}

// ---------------------------------------------------------------------------
extern "C" void kernel_launch(void* const* d_in, const int* in_sizes, int n_in,
                              void* d_out, int out_size, void* d_ws, size_t ws_size,
                              hipStream_t stream) {
  (void)in_sizes; (void)n_in; (void)out_size; (void)ws_size;
  // inputs
  const int* pub   = (const int*)d_in[0];
  const int* priv  = (const int*)d_in[1];
  const float* stacks = (const float*)d_in[2];
  const float* bets   = (const float*)d_in[3];
  const float* act    = (const float*)d_in[4];
  const int* stage    = (const int*)d_in[5];
  const int* cpos     = (const int*)d_in[6];
  const float* amask  = (const float*)d_in[7];
  const float* prev   = (const float*)d_in[8];
  // params
  const float* table = (const float*)d_in[9];
  const float* Wq = (const float*)d_in[10]; const float* bq = (const float*)d_in[11];
  const float* Wk = (const float*)d_in[12]; const float* bk = (const float*)d_in[13];
  const float* Wv = (const float*)d_in[14]; const float* bv = (const float*)d_in[15];
  const float* Wo = (const float*)d_in[16]; const float* bo = (const float*)d_in[17];
  const float* attn_g = (const float*)d_in[18]; const float* attn_b = (const float*)d_in[19];
  const float* st_emb = (const float*)d_in[20]; const float* p_emb = (const float*)d_in[21];
  const float* f1W = (const float*)d_in[22]; const float* f1b = (const float*)d_in[23];
  const float* f1g = (const float*)d_in[24]; const float* f1beta = (const float*)d_in[25];
  const float* f2W = (const float*)d_in[26]; const float* f2b = (const float*)d_in[27];
  const float* f2g = (const float*)d_in[28]; const float* f2beta = (const float*)d_in[29];
  const float* f3W = (const float*)d_in[30]; const float* f3b = (const float*)d_in[31];
  const float* f3g = (const float*)d_in[32]; const float* f3beta = (const float*)d_in[33];
  const float* Wz = (const float*)d_in[34]; const float* bz = (const float*)d_in[35];
  const float* Wr = (const float*)d_in[36]; const float* br = (const float*)d_in[37];
  const float* Wc = (const float*)d_in[38]; const float* bc = (const float*)d_in[39];
  const float* m1W = (const float*)d_in[40]; const float* m1b = (const float*)d_in[41];
  const float* m1skip = (const float*)d_in[42];
  const float* m1g = (const float*)d_in[43]; const float* m1beta = (const float*)d_in[44];
  const float* m2W = (const float*)d_in[45]; const float* m2b = (const float*)d_in[46];
  const float* m2g = (const float*)d_in[47]; const float* m2beta = (const float*)d_in[48];
  const float* hW1 = (const float*)d_in[49]; const float* hb1 = (const float*)d_in[50];
  const float* hW2 = (const float*)d_in[51]; const float* hb2 = (const float*)d_in[52];

  float* outp = (float*)d_out;
  float* ctx_out = outp + (size_t)NB * 10;

  // workspace layout
  char* base = (char*)d_ws;
  size_t off = 0;
  auto alloc = [&](size_t bytes) -> void* {
    void* p = base + off;
    off = (off + bytes + 255) & ~(size_t)255;
    return p;
  };
  bf16* WqT = (bf16*)alloc(8 * 16 * 1024);
  bf16* WkT = (bf16*)alloc(8 * 16 * 1024);
  bf16* WvT = (bf16*)alloc(8 * 16 * 1024);
  bf16* WoT = (bf16*)alloc(8 * 16 * 1024);
  bf16* f1T = (bf16*)alloc((size_t)59 * 64 * 1024);
  bf16* f2T = (bf16*)alloc((size_t)32 * 32 * 1024);
  bf16* f3T = (bf16*)alloc(16 * 16 * 1024);
  bf16* WzT = (bf16*)alloc(16 * 16 * 1024);
  bf16* WrT = (bf16*)alloc(16 * 16 * 1024);
  bf16* WcT = (bf16*)alloc(16 * 16 * 1024);
  bf16* m1T = (bf16*)alloc(16 * 16 * 1024);
  bf16* m2T = (bf16*)alloc(8 * 16 * 1024);
  bf16* h1T = (bf16*)alloc(8 * 32 * 1024);
  bf16* featbf = (bf16*)alloc((size_t)NB * KPADF * 2);
  float* P1 = (float*)alloc((size_t)NB * 1024 * 4);  // pre-LN fp32 scratch
  bf16* Q1 = (bf16*)alloc((size_t)NB * 1024 * 2);    // post-LN bf16 activations
  bf16* comb = (bf16*)alloc((size_t)NB * 512 * 2);   // [feats | gate-input]
  float* zb = (float*)alloc((size_t)NB * 256 * 4);
  float* rb = (float*)alloc((size_t)NB * 256 * 4);
  float* cb = (float*)alloc((size_t)NB * 256 * 4);

  auto prep = [&](const float* src, const float* srcB, int K, int N, int Kpad,
                  bf16* dst, int mode) {
    int Ktiles = Kpad / 32;
    size_t total = (size_t)Ktiles * (N / 16) * 512;
    int blocks = (int)((total + 255) / 256);
    prep_w<<<blocks, 256, 0, stream>>>(src, srcB, K, N, Ktiles, dst, mode);
  };
  auto gemm = [&](const bf16* X, int Kpad, const bf16* WT, const float* bias,
                  const bf16* res, int ldres, float* out, int N, int epi) {
    dim3 g(NB / 64, N / 256);
    gemm_bf16<<<g, 256, 0, stream>>>(X, Kpad, WT, bias, res, ldres, out, N, epi);
  };

  // 1) weight prep (bf16 B-fragment tiling)
  prep(Wq, nullptr, 256, 256, 256, WqT, 0);
  prep(Wk, nullptr, 256, 256, 256, WkT, 0);
  prep(Wv, nullptr, 256, 256, 256, WvT, 0);
  prep(Wo, nullptr, 256, 256, 256, WoT, 0);
  prep(f1W, nullptr, KFEAT, 1024, KPADF, f1T, 0);
  prep(f2W, nullptr, 1024, 512, 1024, f2T, 0);
  prep(f3W, nullptr, 512, 256, 512, f3T, 0);
  prep(Wz, nullptr, 512, 256, 512, WzT, 0);
  prep(Wr, nullptr, 512, 256, 512, WrT, 0);
  prep(Wc, nullptr, 512, 256, 512, WcT, 0);
  prep(m1W, m1skip, 512, 256, 512, m1T, 0);   // fold skip weight: mc@(W+skip)
  prep(m2W, nullptr, 256, 256, 256, m2T, 0);
  prep(hW1, nullptr, 256, 512, 256, h1T, 1);  // all 4 stage heads as 256x512

  // 2) fused embedding + attention + LN -> feat_in[:, 0:1792]
  attn_kernel<<<NB / 4, 256, 0, stream>>>(priv, pub, table, WqT, WkT, WvT, WoT,
                                          bq, bk, bv, bo, attn_g, attn_b, featbf);
  // 3) scalar feature columns + zero pad -> feat_in[:, 1792:1888]
  smallfeat_kernel<<<(NB * 96) / 256, 256, 0, stream>>>(stacks, bets, act, stage,
                                                        cpos, st_emb, p_emb, featbf);
  // 4) trunk MLP
  gemm(featbf, KPADF, f1T, f1b, nullptr, 0, P1, 1024, 0);
  ln_gelu<<<NB, 256, 0, stream>>>(P1, 1024, f1g, f1beta, Q1, 1024);
  gemm(Q1, 1024, f2T, f2b, nullptr, 0, P1, 512, 0);
  ln_gelu<<<NB, 256, 0, stream>>>(P1, 512, f2g, f2beta, Q1, 512);
  gemm(Q1, 512, f3T, f3b, nullptr, 0, P1, 256, 0);
  ln_gelu<<<NB, 256, 0, stream>>>(P1, 256, f3g, f3beta, comb, 512);  // feats -> comb[:,0:256]
  // 5) GRU cell
  cvtprev_kernel<<<NB, 256, 0, stream>>>(prev, comb);                 // comb[:,256:512]=prev
  gemm(comb, 512, WzT, bz, nullptr, 0, zb, 256, 1);                   // sigmoid
  gemm(comb, 512, WrT, br, nullptr, 0, rb, 256, 1);                   // sigmoid
  mulrp_kernel<<<NB, 256, 0, stream>>>(rb, prev, comb);               // comb[:,256:]=r*prev
  gemm(comb, 512, WcT, bc, nullptr, 0, cb, 256, 2);                   // tanh
  gruout_kernel<<<NB, 256, 0, stream>>>(zb, cb, prev, ctx_out, comb); // new_ctx
  // 6) residual MLP blocks
  gemm(comb, 512, m1T, m1b, nullptr, 0, P1, 256, 0);
  ln_gelu<<<NB, 256, 0, stream>>>(P1, 256, m1g, m1beta, Q1, 256);
  gemm(Q1, 256, m2T, m2b, Q1, 256, P1, 256, 0);                       // +residual y
  ln_gelu<<<NB, 256, 0, stream>>>(P1, 256, m2g, m2beta, Q1, 256);
  // 7) per-stage heads
  gemm(Q1, 256, h1T, nullptr, nullptr, 0, P1, 512, 0);                // all 4 stages
  head_kernel<<<NB / 256, 256, 0, stream>>>(P1, stage, hb1, hW2, hb2, amask, outp);
}
